// CausalSelfAttention_54382875902306
// MI455X (gfx1250) — matrix operand order repaired and checked by reading, not compiled, not run on hardware
//
#include <hip/hip_runtime.h>
#include <hip/hip_bf16.h>

// ---------------------------------------------------------------------------
// CausalSelfAttention forward for MI455X (gfx1250, wave32, WMMA)
//   B=4, T=2048, C=1024, H=16, hs=64
// Pipeline:
//   [k_cvt_x, k_wt x2]  one-time f32->f16 conversion + weight transpose
//   [k_qkv]             GEMM 8192x3072x1024, async double-buffered LDS staging
//   [k_attn]            causal flash attention, register-resident fragments
//   [k_proj]            GEMM 8192x1024x1024, async double-buffered LDS staging
// All matmuls via v_wmma_f32_16x16x32_f16 (f16 in, f32 accumulate).
// ---------------------------------------------------------------------------

typedef __attribute__((ext_vector_type(16))) _Float16 v16h;
typedef __attribute__((ext_vector_type(8)))  _Float16 v8h;
typedef __attribute__((ext_vector_type(8)))  float    v8f;
typedef __attribute__((ext_vector_type(4)))  float    v4f;
typedef __attribute__((ext_vector_type(4)))  int      v4i;

#define TT   2048   // sequence length T
#define CC   1024   // channels C
#define NHH  16     // heads
#define HS   64     // head size
#define LDSP 40     // padded LDS row stride in halves (80B, 16B-aligned frag reads)

// gfx1250 async global->LDS copy path (ASYNCcnt), with sync fallback.
#if defined(__gfx1250__) && __has_builtin(__builtin_amdgcn_global_load_async_to_lds_b128) && __has_builtin(__builtin_amdgcn_s_wait_asynccnt)
#define USE_ASYNC 1
typedef __attribute__((address_space(1))) v4i* gp_v4i;  // global int4*
typedef __attribute__((address_space(3))) v4i* lp_v4i;  // LDS int4*
#else
#define USE_ASYNC 0
#endif

__device__ __forceinline__ v16h cat16(v8h lo, v8h hi) {
  return __builtin_shufflevector(lo, hi, 0,1,2,3,4,5,6,7,8,9,10,11,12,13,14,15);
}
__device__ __forceinline__ v8f wmma_f16(v16h a, v16h b, v8f c) {
  // D = A(16x32 f16) x B(32x16 f16) + C(16x16 f32)
  return __builtin_amdgcn_wmma_f32_16x16x32_f16(false, a, false, b, (short)0, c,
                                                false, false);
}
__device__ __forceinline__ v8f zero8() {
  v8f z;
  #pragma unroll
  for (int i = 0; i < 8; ++i) z[i] = 0.0f;
  return z;
}

// Stage a 128-row x 32-half tile (rows stride CC in global) into LDS [row][k].
// 512 x 16B chunks, 2 per thread. Async path: fire-and-forget, caller waits.
__device__ __forceinline__ void stage_tile(const _Float16* __restrict__ g,
                                           _Float16* l, int tid) {
  #pragma unroll
  for (int i = 0; i < 2; ++i) {
    const int idx = tid + 256 * i;
    const int row = idx >> 2, c = idx & 3;
    const _Float16* gp = g + (size_t)row * CC + c * 8;
    _Float16* lp = l + row * LDSP + c * 8;
#if USE_ASYNC
    __builtin_amdgcn_global_load_async_to_lds_b128((gp_v4i)(void*)gp,
                                                   (lp_v4i)(void*)lp, 0, 0);
#else
    *(v8h*)lp = *(const v8h*)gp;
#endif
  }
}

// Load A/B fragments from a staged tile pair and do the 8 WMMAs of one K-step.
__device__ __forceinline__ void tile_mma(const _Float16* lA_, const _Float16* lB_,
                                         int mi, int ni, int lL, int hi,
                                         v8f acc[4][2]) {
  v16h a[4], b[2];
  #pragma unroll
  for (int f = 0; f < 4; ++f) {
    const _Float16* p = lA_ + (mi * 64 + f * 16 + lL) * LDSP + hi * 8;
    a[f] = cat16(*(const v8h*)p, *(const v8h*)(p + 16));
  }
  #pragma unroll
  for (int g = 0; g < 2; ++g) {
    const _Float16* p = lB_ + (ni * 32 + g * 16 + lL) * LDSP + hi * 8;
    b[g] = cat16(*(const v8h*)p, *(const v8h*)(p + 16));
  }
  #pragma unroll
  for (int f = 0; f < 4; ++f)
    #pragma unroll
    for (int g = 0; g < 2; ++g)
      acc[f][g] = wmma_f16(a[f], b[g], acc[f][g]);
}

// ---------------------------------------------------------------------------
// Kernel 0a: x (f32) -> xh (f16), flat. 8 elements/thread.
// ---------------------------------------------------------------------------
__global__ __launch_bounds__(256) void k_cvt_x(const float* __restrict__ in,
                                               _Float16* __restrict__ outh) {
  const size_t base = ((size_t)blockIdx.x * 256 + threadIdx.x) * 8;
  v4f a = *(const v4f*)(in + base);
  v4f b = *(const v4f*)(in + base + 4);
  v8h h;
  #pragma unroll
  for (int j = 0; j < 4; ++j) { h[j] = (_Float16)a[j]; h[4 + j] = (_Float16)b[j]; }
  *(v8h*)(outh + base) = h;
}

// ---------------------------------------------------------------------------
// Kernel 0b: W (f32, [1024][N]) -> Wt (f16, [N][1024]) via 64x64 LDS tiles.
// ---------------------------------------------------------------------------
__global__ __launch_bounds__(256) void k_wt(const float* __restrict__ W,
                                            _Float16* __restrict__ Wt, int N) {
  __shared__ _Float16 t[64 * 80];
  const int tid = threadIdx.x;
  const int n0 = blockIdx.x * 64, k0 = blockIdx.y * 64;
  #pragma unroll
  for (int i = 0; i < 4; ++i) {
    const int idx = tid + 256 * i;          // 1024 float4 chunks
    const int kr = idx >> 4, c4 = idx & 15; // k row, n chunk
    v4f v = *(const v4f*)(W + (size_t)(k0 + kr) * N + n0 + c4 * 4);
    #pragma unroll
    for (int j = 0; j < 4; ++j)
      t[(c4 * 4 + j) * 80 + kr] = (_Float16)v[j];
  }
  __syncthreads();
  #pragma unroll
  for (int i = 0; i < 2; ++i) {
    const int idx = tid + 256 * i;          // 512 v8h chunks
    const int nr = idx >> 3, c = idx & 7;
    *(v8h*)(Wt + (size_t)(n0 + nr) * 1024 + k0 + c * 8) =
        *(const v8h*)(t + nr * 80 + c * 8);
  }
}

// ---------------------------------------------------------------------------
// Kernel 1: qkv = xh @ Wt_attn^T + b_attn  (M=8192, N=3072, K=1024)
//   Q (pre-scaled 0.125) -> [BH,T,HS]; K -> [BH,T,HS]; V^T -> [BH,HS,T]  (f16)
// Double-buffered async LDS staging; 8 WMMAs per K-step of 32.
// ---------------------------------------------------------------------------
__global__ __launch_bounds__(256) void k_qkv(
    const _Float16* __restrict__ xh, const _Float16* __restrict__ Wt,
    const float* __restrict__ bias,
    _Float16* __restrict__ Qb, _Float16* __restrict__ Kb,
    _Float16* __restrict__ Vt)
{
  __shared__ _Float16 lA[2][128 * LDSP];
  __shared__ _Float16 lB[2][128 * LDSP];

  const int tid = threadIdx.x;
  const int lane = tid & 31, w = tid >> 5;
  const int lL = lane & 15, hi = lane >> 4;
  const int mi = w >> 2, ni = w & 3;
  const int n0 = blockIdx.x * 128, m0 = blockIdx.y * 128;

  const _Float16* Ab = xh + (size_t)m0 * CC;  // rows m0.., stride CC
  const _Float16* Bb = Wt + (size_t)n0 * CC;  // rows n0.., stride CC

  v8f acc[4][2];
  #pragma unroll
  for (int f = 0; f < 4; ++f)
    #pragma unroll
    for (int g = 0; g < 2; ++g) acc[f][g] = zero8();

  stage_tile(Ab, lA[0], tid);                 // prologue: tile 0 in flight
  stage_tile(Bb, lB[0], tid);

  const int S = CC / 32;
  for (int s = 0; s < S; ++s) {
    const int p = s & 1;
    if (s + 1 < S) {
      stage_tile(Ab + (s + 1) * 32, lA[p ^ 1], tid);
      stage_tile(Bb + (s + 1) * 32, lB[p ^ 1], tid);
#if USE_ASYNC
      __builtin_amdgcn_s_wait_asynccnt(4);    // tile s done (in-order), s+1 in flight
#endif
    } else {
#if USE_ASYNC
      __builtin_amdgcn_s_wait_asynccnt(0);
#endif
    }
    __syncthreads();
    tile_mma(lA[p], lB[p], mi, ni, lL, hi, acc);
    __syncthreads();
  }

  // ---- epilogue: bias + scatter into Q/K/V^T head layouts
  #pragma unroll
  for (int f = 0; f < 4; ++f) {
    #pragma unroll
    for (int g = 0; g < 2; ++g) {
      const int col = n0 + ni * 32 + g * 16 + lL;
      const float bs = bias[col];
      const int seg = col >> 10, cn = col & (CC - 1);
      const int h = cn >> 6, d = cn & (HS - 1);
      #pragma unroll
      for (int r = 0; r < 8; ++r) {
        const int row = m0 + mi * 64 + f * 16 + r + hi * 8;
        const int bb = row >> 11, t = row & (TT - 1);
        const int bh = bb * NHH + h;
        const float val = acc[f][g][r] + bs;
        if (seg == 0)
          Qb[((size_t)bh * TT + t) * HS + d] = (_Float16)(val * 0.125f);
        else if (seg == 1)
          Kb[((size_t)bh * TT + t) * HS + d] = (_Float16)val;
        else
          Vt[((size_t)bh * HS + d) * TT + t] = (_Float16)val;
      }
    }
  }
}

// ---------------------------------------------------------------------------
// Kernel 2: causal flash attention. 8 independent waves/block, 16 q-rows each.
// S = Q K^T (4 WMMA), online softmax, O += P V (4 WMMA) per 32-key chunk.
// K/V B-fragments load straight from global into WMMA register layout.
// ---------------------------------------------------------------------------
__global__ __launch_bounds__(256) void k_attn(
    const _Float16* __restrict__ Qb, const _Float16* __restrict__ Kb,
    const _Float16* __restrict__ Vt, _Float16* __restrict__ Yb)
{
  __shared__ _Float16 lP[8 * 16 * LDSP];
  const int tid = threadIdx.x, lane = tid & 31, w = tid >> 5;
  const int lL = lane & 15, hi = lane >> 4;
  const int bh = blockIdx.y;
  const int q0 = blockIdx.x * 128 + w * 16;

  const _Float16* Qh = Qb + (size_t)bh * TT * HS;
  const _Float16* Kh = Kb + (size_t)bh * TT * HS;
  const _Float16* Vh = Vt + (size_t)bh * HS * TT;
  _Float16* lp = lP + w * 16 * LDSP;

  v16h aq[2];
  #pragma unroll
  for (int i = 0; i < 2; ++i) {
    const _Float16* p = Qh + (size_t)(q0 + lL) * HS + i * 32 + hi * 8;
    aq[i] = cat16(*(const v8h*)p, *(const v8h*)(p + 16));
  }

  v8f o[4];
  float mr[8], lr[8];
  #pragma unroll
  for (int j = 0; j < 4; ++j) o[j] = zero8();
  #pragma unroll
  for (int r = 0; r < 8; ++r) { mr[r] = -1e30f; lr[r] = 0.0f; }

  const float L2E = 1.44269504088896f;

  for (int k0 = 0; k0 < q0 + 16; k0 += 32) {
    v8f s[2]; s[0] = zero8(); s[1] = zero8();
    #pragma unroll
    for (int dd = 0; dd < 2; ++dd) {
      #pragma unroll
      for (int g = 0; g < 2; ++g) {
        int key = k0 + g * 16 + lL;
        if (key > TT - 1) key = TT - 1;        // clamp (masked later)
        const _Float16* p = Kh + (size_t)key * HS + dd * 32 + hi * 8;
        v16h kb = cat16(*(const v8h*)p, *(const v8h*)(p + 16));
        s[g] = wmma_f16(aq[dd], kb, s[g]);
      }
    }

    #pragma unroll
    for (int r = 0; r < 8; ++r) {
      const int qrow = q0 + r + hi * 8;
      #pragma unroll
      for (int g = 0; g < 2; ++g) {
        const int key = k0 + g * 16 + lL;
        if (key > qrow) s[g][r] = -1e30f;
      }
      float mx = fmaxf(s[0][r], s[1][r]);
      #pragma unroll
      for (int off = 1; off < 16; off <<= 1) mx = fmaxf(mx, __shfl_xor(mx, off));
      const float nm = fmaxf(mr[r], mx);
      const float al = __builtin_exp2f((mr[r] - nm) * L2E);
      s[0][r] = __builtin_exp2f((s[0][r] - nm) * L2E);
      s[1][r] = __builtin_exp2f((s[1][r] - nm) * L2E);
      float rs = s[0][r] + s[1][r];
      #pragma unroll
      for (int off = 1; off < 16; off <<= 1) rs += __shfl_xor(rs, off);
      lr[r] = lr[r] * al + rs;
      mr[r] = nm;
      #pragma unroll
      for (int j = 0; j < 4; ++j) o[j][r] *= al;
      lp[(r + hi * 8) * LDSP + lL]      = (_Float16)s[0][r];
      lp[(r + hi * 8) * LDSP + 16 + lL] = (_Float16)s[1][r];
    }

    const _Float16* pp = lp + lL * LDSP + hi * 16;
    v16h pa = cat16(*(const v8h*)pp, *(const v8h*)(pp + 8));
    int kb2 = k0 + hi * 16;
    if (kb2 > TT - 16) kb2 = TT - 16;          // clamp (p==0 there)
    #pragma unroll
    for (int j = 0; j < 4; ++j) {
      const _Float16* p = Vh + (size_t)(j * 16 + lL) * TT + kb2;
      v16h vb = cat16(*(const v8h*)p, *(const v8h*)(p + 8));
      o[j] = wmma_f16(pa, vb, o[j]);
    }
  }

  const int bb = bh >> 4, h = bh & (NHH - 1);
  #pragma unroll
  for (int r = 0; r < 8; ++r) {
    const float inv = 1.0f / lr[r];
    const int q = q0 + r + hi * 8;
    const size_t yrow = ((size_t)bb * TT + q) * CC + h * HS;
    #pragma unroll
    for (int j = 0; j < 4; ++j)
      Yb[yrow + j * 16 + lL] = (_Float16)(o[j][r] * inv);
  }
}

// ---------------------------------------------------------------------------
// Kernel 3: out = y @ Wt_proj^T + b_proj  (M=8192, N=1024, K=1024), f32 out
// ---------------------------------------------------------------------------
__global__ __launch_bounds__(256) void k_proj(
    const _Float16* __restrict__ Yb, const _Float16* __restrict__ Wt,
    const float* __restrict__ bias, float* __restrict__ out)
{
  __shared__ _Float16 lA[2][128 * LDSP];
  __shared__ _Float16 lB[2][128 * LDSP];

  const int tid = threadIdx.x;
  const int lane = tid & 31, w = tid >> 5;
  const int lL = lane & 15, hi = lane >> 4;
  const int mi = w >> 2, ni = w & 3;
  const int n0 = blockIdx.x * 128, m0 = blockIdx.y * 128;

  const _Float16* Ab = Yb + (size_t)m0 * CC;
  const _Float16* Bb = Wt + (size_t)n0 * CC;

  v8f acc[4][2];
  #pragma unroll
  for (int f = 0; f < 4; ++f)
    #pragma unroll
    for (int g = 0; g < 2; ++g) acc[f][g] = zero8();

  stage_tile(Ab, lA[0], tid);
  stage_tile(Bb, lB[0], tid);

  const int S = CC / 32;
  for (int s = 0; s < S; ++s) {
    const int p = s & 1;
    if (s + 1 < S) {
      stage_tile(Ab + (s + 1) * 32, lA[p ^ 1], tid);
      stage_tile(Bb + (s + 1) * 32, lB[p ^ 1], tid);
#if USE_ASYNC
      __builtin_amdgcn_s_wait_asynccnt(4);
#endif
    } else {
#if USE_ASYNC
      __builtin_amdgcn_s_wait_asynccnt(0);
#endif
    }
    __syncthreads();
    tile_mma(lA[p], lB[p], mi, ni, lL, hi, acc);
    __syncthreads();
  }

  #pragma unroll
  for (int f = 0; f < 4; ++f) {
    #pragma unroll
    for (int g = 0; g < 2; ++g) {
      const int col = n0 + ni * 32 + g * 16 + lL;
      const float bs = bias[col];
      #pragma unroll
      for (int r = 0; r < 8; ++r) {
        const int row = m0 + mi * 64 + f * 16 + r + hi * 8;
        out[(size_t)row * CC + col] = acc[f][g][r] + bs;
      }
    }
  }
}

// ---------------------------------------------------------------------------
extern "C" void kernel_launch(void* const* d_in, const int* in_sizes, int n_in,
                              void* d_out, int out_size, void* d_ws, size_t ws_size,
                              hipStream_t stream) {
  const float* x      = (const float*)d_in[0];
  const float* W_attn = (const float*)d_in[1];
  const float* b_attn = (const float*)d_in[2];
  const float* W_proj = (const float*)d_in[3];
  const float* b_proj = (const float*)d_in[4];

  char* ws = (char*)d_ws;
  const size_t seg = (size_t)4 * NHH * TT * HS * sizeof(_Float16); // 16.78 MB
  _Float16* Qb = (_Float16*)(ws);
  _Float16* Kb = (_Float16*)(ws + seg);
  _Float16* Vt = (_Float16*)(ws + 2 * seg);
  _Float16* Yb = (_Float16*)(ws + 3 * seg);
  _Float16* xh = (_Float16*)(ws + 4 * seg);
  _Float16* Wta = (_Float16*)(ws + 5 * seg);                       // 6.29 MB
  _Float16* Wtp = (_Float16*)(ws + 5 * seg + (size_t)3 * CC * CC * 2);

  // 0) one-time conversions: x -> f16, W -> f16 transposed [N][K]
  k_cvt_x<<<dim3((4 * TT * CC) / (256 * 8)), 256, 0, stream>>>(x, xh);
  k_wt<<<dim3(3 * CC / 64, CC / 64), 256, 0, stream>>>(W_attn, Wta, 3 * CC);
  k_wt<<<dim3(CC / 64, CC / 64), 256, 0, stream>>>(W_proj, Wtp, CC);

  // 1) QKV projection
  k_qkv<<<dim3(3 * CC / 128, 4 * TT / 128), 256, 0, stream>>>(
      xh, Wta, b_attn, Qb, Kb, Vt);

  // 2) causal flash attention
  k_attn<<<dim3(TT / 128, 4 * NHH), 256, 0, stream>>>(Qb, Kb, Vt, Yb);

  // 3) output projection
  k_proj<<<dim3(CC / 128, 4 * TT / 128), 256, 0, stream>>>(
      Yb, Wtp, b_proj, (float*)d_out);
}
